// ConsciousSwarm_70815420776831
// MI455X (gfx1250) — matrix operand order repaired and checked by reading, not compiled, not run on hardware
//
#include <hip/hip_runtime.h>
#include <math.h>

typedef __attribute__((ext_vector_type(16))) _Float16 v16h;
typedef __attribute__((ext_vector_type(8)))  float    v8f;

#define BB 8
#define TT 256
#define NN 32
#define DD 128
#define SS 8
#define VV 32000
#define HH 4
#define HD 32
#define NCH 32
#define CH 8

__device__ __forceinline__ float geluf(float x){
  return 0.5f * x * (1.0f + erff(x * 0.70710678118654752f));
}
__device__ __forceinline__ float sigmf(float x){
  return 1.0f / (1.0f + expf(-x));
}
__device__ __forceinline__ float wred(float v){
  #pragma unroll
  for (int off = 16; off > 0; off >>= 1) v += __shfl_xor(v, off, 32);
  return v;
}

// ---------------------------------------------------------------- f32 -> f16
__global__ void k_f32_to_f16(const float* __restrict__ src, _Float16* __restrict__ dst, int n){
  int i = blockIdx.x * blockDim.x + threadIdx.x;
  int stride = gridDim.x * blockDim.x;
  for (; i < n; i += stride) dst[i] = (_Float16)src[i];
}

// ------------------------------------------------------- percepts (embed+ifc)
__global__ __launch_bounds__(128) void k_percepts(
    const int* __restrict__ idx, const float* __restrict__ emb,
    const float* __restrict__ pos, const float* __restrict__ w,
    const float* __restrict__ bvec, const float* __restrict__ gam,
    const float* __restrict__ bet, float* __restrict__ outp){
  int row = blockIdx.x;          // b*T + t
  int t = row % TT;
  int d = threadIdx.x;
  __shared__ float s_raw[DD];
  __shared__ float s_red[16];
  int tok = idx[row];
  s_raw[d] = emb[(size_t)tok*DD + d] + pos[t*DD + d];
  __syncthreads();
  float acc = bvec[d];
  for (int i = 0; i < DD; i++) acc += s_raw[i] * w[i*DD + d];
  float x = geluf(acc);
  float sv = wred(x), sq = wred(x*x);
  int wi = d >> 5;
  if ((d & 31) == 0){ s_red[wi] = sv; s_red[8+wi] = sq; }
  __syncthreads();
  float s  = s_red[0]+s_red[1]+s_red[2]+s_red[3];
  float ss = s_red[8]+s_red[9]+s_red[10]+s_red[11];
  float m = s * (1.0f/DD);
  float inv = rsqrtf(ss*(1.0f/DD) - m*m + 1e-5f);
  outp[(size_t)row*DD + d] = (x - m)*inv*gam[d] + bet[d];
}

// ----------------------------------------------------- persistent recurrence
struct RecArgs {
  const float* percepts;
  const float *sproj_w, *sproj_b, *sgate_w, *sgate_b;
  const float *sal_w, *sal_b;
  const float *bcast_w, *bcast_b, *wln_g, *wln_b;
  const float *res_w, *res_b, *out_w, *out_b;
  const float *cq_w, *cq_b, *ck_w, *ck_b, *cv_w, *cv_b, *co_w, *co_b;
  const float *cln_g, *cln_b;
  const float *P_w, *P_b, *D1_w, *D1_b, *D2_w, *D2_b;
  const float *G_w, *G_b, *SM_w, *SM_b, *PR_w, *PR_b;
  const float *ln_g, *ln_b;
  _Float16* proj_h;
  float *gQ, *gK, *gV, *gMsg, *gPrev;
};

__device__ __forceinline__ void row_ln_1024(const float* src, float* dst,
                                            const float* gam, const float* bet, int tid){
  int n = tid >> 5, l = tid & 31;           // 32 waves <-> 32 node rows
  float x0 = src[n*DD+l],    x1 = src[n*DD+l+32];
  float x2 = src[n*DD+l+64], x3 = src[n*DD+l+96];
  float s  = wred(x0+x1+x2+x3);
  float ss = wred(x0*x0+x1*x1+x2*x2+x3*x3);
  float m = s*(1.0f/DD);
  float inv = rsqrtf(ss*(1.0f/DD) - m*m + 1e-5f);
  dst[n*DD+l]    = (x0-m)*inv*gam[l]    + bet[l];
  dst[n*DD+l+32] = (x1-m)*inv*gam[l+32] + bet[l+32];
  dst[n*DD+l+64] = (x2-m)*inv*gam[l+64] + bet[l+64];
  dst[n*DD+l+96] = (x3-m)*inv*gam[l+96] + bet[l+96];
}

__global__ __launch_bounds__(1024) void k_recurrence(RecArgs A){
  const int tid = threadIdx.x;
  const int b = blockIdx.x;
  const int bbase = b * NN * DD;

  __shared__ float s_states[NN*DD];   // 16 KB
  __shared__ float s_tmp[NN*DD];      // 16 KB
  __shared__ float s_sens[SS*DD];     // 4 KB
  __shared__ float s_sg[SS*DD];       // 4 KB
  __shared__ float s_cur[DD], s_ws[DD], s_res[DD], s_bcast[DD], s_inc[DD];
  __shared__ float s_sal[NN], s_sur[NN], s_red[16];

  for (int k = 0; k < 4; k++) s_states[tid + (k<<10)] = 0.0f;
  __syncthreads();

  for (int c = 0; c < NCH; c++){
    // ================= 8 timesteps =================
    for (int t = 0; t < CH; t++){
      const int row = b*TT + c*CH + t;
      if (tid < DD) s_cur[tid] = A.percepts[(size_t)row*DD + tid];
      __syncthreads();
      { // sensory = cur @ sproj_w + b   (1024 outputs)
        float acc = A.sproj_b[tid];
        for (int i = 0; i < DD; i++) acc += s_cur[i] * A.sproj_w[i*(SS*DD) + tid];
        s_sens[tid] = acc;
      }
      __syncthreads();
      { // sg = sigmoid([cs, sensory] @ sgate_w + b)
        int s = tid >> 7, d = tid & 127;
        const float* cs = s_states + s*DD;
        const float* sn = s_sens  + s*DD;
        float acc = A.sgate_b[d];
        for (int i = 0; i < DD; i++){
          acc += cs[i]*A.sgate_w[i*DD + d];
          acc += sn[i]*A.sgate_w[(DD+i)*DD + d];
        }
        s_sg[tid] = sigmf(acc);
      }
      __syncthreads();
      { // states[:, :S] = sg*cs + (1-sg)*sensory
        float g = s_sg[tid];
        s_states[tid] = g*s_states[tid] + (1.0f - g)*s_sens[tid];
      }
      __syncthreads();
      { // salience, one wave per node
        int n = tid >> 5, l = tid & 31;
        const float* sr = s_states + n*DD;
        float p = sr[l]*A.sal_w[l] + sr[l+32]*A.sal_w[l+32]
                + sr[l+64]*A.sal_w[l+64] + sr[l+96]*A.sal_w[l+96];
        p = wred(p);
        if (l == 0) s_sal[n] = p + A.sal_b[0];
      }
      __syncthreads();
      if (tid < NN){ // softmax(sal*5) in a single wave
        float x = s_sal[tid]*5.0f;
        float mx = x;
        #pragma unroll
        for (int off = 16; off > 0; off >>= 1) mx = fmaxf(mx, __shfl_xor(mx, off, 32));
        float e = expf(x - mx);
        float sm = wred(e);
        s_sal[tid] = e / sm;
      }
      __syncthreads();
      if (tid < DD){ // workspace
        float acc = 0.0f;
        for (int n = 0; n < NN; n++) acc += s_states[n*DD + tid]*s_sal[n];
        s_ws[tid] = acc;
      } else if (tid < 2*DD){ // residual
        int d = tid - DD;
        float acc = A.res_b[d];
        for (int i = 0; i < DD; i++) acc += s_cur[i]*A.res_w[i*DD + d];
        s_res[d] = acc;
      }
      __syncthreads();
      if (tid < DD){ // proj = gelu([ws, res] @ out_w + b)  -> f16 for WMMA GEMM
        float acc = A.out_b[tid];
        for (int i = 0; i < DD; i++){
          acc += s_ws[i]*A.out_w[i*DD + tid];
          acc += s_res[i]*A.out_w[(DD+i)*DD + tid];
        }
        A.proj_h[(size_t)row*DD + tid] = (_Float16)geluf(acc);
      } else if (tid < 2*DD){ // bcast pre-LN (into s_sg[0:128], free now)
        int d = tid - DD;
        float acc = A.bcast_b[d];
        for (int i = 0; i < DD; i++) acc += s_ws[i]*A.bcast_w[i*DD + d];
        s_sg[d] = acc;
      }
      __syncthreads();
      { // LN(bcast) with wln
        float v = (tid < DD) ? s_sg[tid] : 0.0f;
        float sv = wred(v), sq = wred(v*v);
        if (tid < DD && (tid & 31) == 0){ s_red[tid>>5] = sv; s_red[8+(tid>>5)] = sq; }
        __syncthreads();
        if (tid < DD){
          float s  = s_red[0]+s_red[1]+s_red[2]+s_red[3];
          float ss = s_red[8]+s_red[9]+s_red[10]+s_red[11];
          float m = s*(1.0f/DD);
          float inv = rsqrtf(ss*(1.0f/DD) - m*m + 1e-5f);
          s_bcast[tid] = (v - m)*inv*A.wln_g[tid] + A.wln_b[tid];
        }
        __syncthreads();
      }
    } // t

    // ================= chunk update =================
    for (int k = 0; k < 4; k++){ // Q, K, V projections
      int o = tid + (k<<10);
      int n = o >> 7, d = o & 127;
      const float* sr = s_states + n*DD;
      float aq = A.cq_b[d], ak = A.ck_b[d], av = A.cv_b[d];
      for (int i = 0; i < DD; i++){
        float sv = sr[i];
        aq += sv*A.cq_w[i*DD+d];
        ak += sv*A.ck_w[i*DD+d];
        av += sv*A.cv_w[i*DD+d];
      }
      A.gQ[bbase+o] = aq; A.gK[bbase+o] = ak; A.gV[bbase+o] = av;
    }
    __syncthreads();
    if (tid < HH*NN){ // attention: thread = (head, query-node)
      int h = tid >> 5, i = tid & 31;
      const float* qp = A.gQ + bbase + i*DD + h*HD;
      float q[HD];
      #pragma unroll
      for (int k = 0; k < HD; k++) q[k] = qp[k];
      float sc[NN]; float mx = -1e30f;
      #pragma unroll
      for (int j = 0; j < NN; j++){
        const float* kp = A.gK + bbase + j*DD + h*HD;
        float s = 0.0f;
        #pragma unroll
        for (int k = 0; k < HD; k++) s += q[k]*kp[k];
        s *= 0.17677669529663689f;   // 1/sqrt(32)
        sc[j] = s; mx = fmaxf(mx, s);
      }
      float sum = 0.0f;
      #pragma unroll
      for (int j = 0; j < NN; j++){ sc[j] = expf(sc[j]-mx); sum += sc[j]; }
      float inv = 1.0f/sum;
      float mg[HD];
      #pragma unroll
      for (int k = 0; k < HD; k++) mg[k] = 0.0f;
      #pragma unroll
      for (int j = 0; j < NN; j++){
        float p = sc[j]*inv;
        const float* vp = A.gV + bbase + j*DD + h*HD;
        #pragma unroll
        for (int k = 0; k < HD; k++) mg[k] += p*vp[k];
      }
      float* mp = A.gMsg + bbase + i*DD + h*HD;
      #pragma unroll
      for (int k = 0; k < HD; k++) mp[k] = mg[k];
    }
    __syncthreads();
    for (int k = 0; k < 4; k++){ // states + msg@co -> s_tmp
      int o = tid + (k<<10);
      int n = o >> 7, d = o & 127;
      const float* mr = A.gMsg + bbase + n*DD;
      float acc = A.co_b[d];
      for (int i = 0; i < DD; i++) acc += mr[i]*A.co_w[i*DD+d];
      s_tmp[o] = s_states[o] + acc;
    }
    __syncthreads();
    row_ln_1024(s_tmp, s_states, A.cln_g, A.cln_b, tid);
    __syncthreads();
    if (tid < DD){ // incoming mean over nodes
      float acc = 0.0f;
      for (int n = 0; n < NN; n++) acc += s_states[n*DD + tid];
      s_inc[tid] = acc*(1.0f/NN);
    }
    __syncthreads();
    { // surprise (one wave per node)
      int n = tid >> 5, l = tid & 31;
      const float* pp = A.gPrev + bbase + n*DD;
      float p = 0.0f;
      #pragma unroll
      for (int k = 0; k < 4; k++){
        int d = l + 32*k;
        float df = s_inc[d] - pp[d];
        p += df*df;
      }
      p = wred(p) * (1.0f/DD);
      if (l == 0) s_sur[n] = (c > 0) ? sigmf(p*10.0f) : 0.0f;
    }
    __syncthreads();
    for (int k = 0; k < 4; k++){ // perception -> gQ, self_pred -> gK
      int o = tid + (k<<10);
      int n = o >> 7, d = o & 127;
      float scale = 1.0f + 0.5f*s_sur[n];
      const float* pw = A.P_w + (size_t)n*(3*DD*DD);
      const float* sr = s_states + n*DD;
      float acc = A.P_b[n*DD + d];
      for (int i = 0; i < DD; i++) acc += (s_inc[i]*scale)*pw[i*DD + d];
      for (int i = 0; i < DD; i++) acc += s_bcast[i]*pw[(DD+i)*DD + d];
      for (int i = 0; i < DD; i++) acc += sr[i]*pw[(2*DD+i)*DD + d];
      A.gQ[bbase+o] = geluf(acc);
      const float* smw = A.SM_w + (size_t)n*(DD*DD);
      float ac2 = A.SM_b[n*DD + d];
      for (int i = 0; i < DD; i++) ac2 += sr[i]*smw[i*DD + d];
      A.gK[bbase+o] = ac2;
    }
    __syncthreads();
    for (int k = 0; k < 4; k++){ // d1 = gelu([perc, sp] @ D1) -> gV
      int o = tid + (k<<10);
      int n = o >> 7, d = o & 127;
      const float* w  = A.D1_w + (size_t)n*(2*DD*DD);
      const float* pr = A.gQ + bbase + n*DD;
      const float* sp = A.gK + bbase + n*DD;
      float acc = A.D1_b[n*DD + d];
      for (int i = 0; i < DD; i++) acc += pr[i]*w[i*DD+d] + sp[i]*w[(DD+i)*DD+d];
      A.gV[bbase+o] = geluf(acc);
    }
    __syncthreads();
    for (int k = 0; k < 4; k++){ // decision = d1 @ D2 -> gMsg
      int o = tid + (k<<10);
      int n = o >> 7, d = o & 127;
      const float* w  = A.D2_w + (size_t)n*(DD*DD);
      const float* d1 = A.gV + bbase + n*DD;
      float acc = A.D2_b[n*DD + d];
      for (int i = 0; i < DD; i++) acc += d1[i]*w[i*DD+d];
      A.gMsg[bbase+o] = acc;
    }
    __syncthreads();
    for (int k = 0; k < 4; k++){ // g gate, blend -> s_tmp
      int o = tid + (k<<10);
      int n = o >> 7, d = o & 127;
      const float* w   = A.G_w + (size_t)n*(2*DD*DD);
      const float* sr  = s_states + n*DD;
      const float* dec = A.gMsg + bbase + n*DD;
      float acc = A.G_b[n*DD + d];
      for (int i = 0; i < DD; i++) acc += sr[i]*w[i*DD+d] + dec[i]*w[(DD+i)*DD+d];
      float g = sigmf(acc);
      s_tmp[o] = g*dec[d] + (1.0f - g)*s_states[o];
    }
    __syncthreads();
    row_ln_1024(s_tmp, s_states, A.ln_g, A.ln_b, tid);
    __syncthreads();
    for (int k = 0; k < 4; k++){ // prediction -> gPrev (for next chunk's surprise)
      int o = tid + (k<<10);
      int n = o >> 7, d = o & 127;
      const float* w  = A.PR_w + (size_t)n*(DD*DD);
      const float* sr = s_states + n*DD;
      float acc = A.PR_b[n*DD + d];
      for (int i = 0; i < DD; i++) acc += sr[i]*w[i*DD+d];
      A.gPrev[bbase+o] = acc;
    }
    __syncthreads();
  } // c
}

// --------------------------------------------- logits GEMM: WMMA f16 -> f32
// C(2048 x 32000) = proj_h(2048 x 128) @ emb_h(32000 x 128)^T + bias
__global__ __launch_bounds__(128) void k_logits(
    const _Float16* __restrict__ Ah,   // (M=2048, K=128) row-major
    const _Float16* __restrict__ Bh,   // emb_h (V, K) row-major (acts as B^T)
    const float* __restrict__ bias, float* __restrict__ out){
  const int wave = threadIdx.x >> 5;
  const int lane = threadIdx.x & 31;
  const int g  = lane >> 4;       // half-wave group
  const int hl = lane & 15;
  const int m0 = blockIdx.y * 16;
  const int n0 = blockIdx.x * 256 + wave * 64;

  v8f acc[4] = {};
  for (int k0 = 0; k0 < DD; k0 += 32){
    // A fragment (16x32): lane holds row m0+hl, K = k0+16g+i
    v16h a;
    const _Float16* arow = Ah + (size_t)(m0 + hl)*DD + k0 + 16*g;
    #pragma unroll
    for (int i = 0; i < 8; i++) a[i]   = arow[i];
    #pragma unroll
    for (int i = 0; i < 8; i++) a[8+i] = arow[8 + i];
    #pragma unroll
    for (int t = 0; t < 4; t++){
      int col = n0 + t*16 + hl;
      // B fragment (32x16): lane holds column `col`, K = k0+16g+i -> contiguous in emb row
      const _Float16* brow = Bh + (size_t)col*DD + k0 + 16*g;
      v16h bf;
      #pragma unroll
      for (int i = 0; i < 16; i++) bf[i] = brow[i];
      acc[t] = __builtin_amdgcn_wmma_f32_16x16x32_f16(
          false, a, false, bf, (short)0, acc[t], false, false);
    }
  }
  #pragma unroll
  for (int t = 0; t < 4; t++){
    int col = n0 + t*16 + hl;
    float bv = bias[col];
    #pragma unroll
    for (int j = 0; j < 8; j++){
      int row = m0 + j + 8*g;            // C/D layout: VGPR j = row j (g=0) / 8+j (g=1)
      out[(size_t)row*VV + col] = acc[t][j] + bv;
    }
  }
}

// ---------------------------------------------------------------------- host
extern "C" void kernel_launch(void* const* d_in, const int* in_sizes, int n_in,
                              void* d_out, int out_size, void* d_ws, size_t ws_size,
                              hipStream_t stream) {
  (void)in_sizes; (void)n_in; (void)out_size; (void)ws_size;
  const int*   idx      = (const int*)  d_in[0];
  const float* emb      = (const float*)d_in[1];
  const float* pos_emb  = (const float*)d_in[2];
  const float* ifc_w    = (const float*)d_in[3];
  const float* ifc_b    = (const float*)d_in[4];
  const float* ifc_g    = (const float*)d_in[5];
  const float* ifc_beta = (const float*)d_in[6];

  RecArgs A;
  A.sproj_w = (const float*)d_in[7];   A.sproj_b = (const float*)d_in[8];
  A.sgate_w = (const float*)d_in[9];   A.sgate_b = (const float*)d_in[10];
  A.P_w  = (const float*)d_in[11];  A.P_b  = (const float*)d_in[12];
  A.D1_w = (const float*)d_in[13];  A.D1_b = (const float*)d_in[14];
  A.D2_w = (const float*)d_in[15];  A.D2_b = (const float*)d_in[16];
  // d_in[17], d_in[18] = A_w, A_b : unused by the reference graph
  A.G_w  = (const float*)d_in[19];  A.G_b  = (const float*)d_in[20];
  A.SM_w = (const float*)d_in[21];  A.SM_b = (const float*)d_in[22];
  A.PR_w = (const float*)d_in[23];  A.PR_b = (const float*)d_in[24];
  A.cq_w = (const float*)d_in[25];  A.cq_b = (const float*)d_in[26];
  A.ck_w = (const float*)d_in[27];  A.ck_b = (const float*)d_in[28];
  A.cv_w = (const float*)d_in[29];  A.cv_b = (const float*)d_in[30];
  A.co_w = (const float*)d_in[31];  A.co_b = (const float*)d_in[32];
  A.cln_g = (const float*)d_in[33]; A.cln_b = (const float*)d_in[34];
  A.sal_w = (const float*)d_in[35]; A.sal_b = (const float*)d_in[36];
  A.bcast_w = (const float*)d_in[37]; A.bcast_b = (const float*)d_in[38];
  A.wln_g = (const float*)d_in[39]; A.wln_b = (const float*)d_in[40];
  A.res_w = (const float*)d_in[41]; A.res_b = (const float*)d_in[42];
  A.out_w = (const float*)d_in[43]; A.out_b = (const float*)d_in[44];
  const float* out_bias = (const float*)d_in[45];
  A.ln_g = (const float*)d_in[46];  A.ln_b = (const float*)d_in[47];

  // workspace layout (all 256B aligned)
  char* ws = (char*)d_ws;
  constexpr size_t OFF_EMBH = 0;                                   // V*D f16 = 8,192,000
  constexpr size_t OFF_PERC = OFF_EMBH + (size_t)VV*DD*2;          // B*T*D f32
  constexpr size_t OFF_PROJ = OFF_PERC + (size_t)BB*TT*DD*4;       // B*T*D f16
  constexpr size_t OFF_GBUF = OFF_PROJ + (size_t)BB*TT*DD*2;       // 5 x B*N*D f32
  _Float16* emb_h    = (_Float16*)(ws + OFF_EMBH);
  float*    percepts = (float*)   (ws + OFF_PERC);
  _Float16* proj_h   = (_Float16*)(ws + OFF_PROJ);
  float*    gbuf     = (float*)   (ws + OFF_GBUF);
  A.percepts = percepts;
  A.proj_h   = proj_h;
  A.gQ    = gbuf;
  A.gK    = gbuf + 1*(size_t)BB*NN*DD;
  A.gV    = gbuf + 2*(size_t)BB*NN*DD;
  A.gMsg  = gbuf + 3*(size_t)BB*NN*DD;
  A.gPrev = gbuf + 4*(size_t)BB*NN*DD;

  k_f32_to_f16<<<4096, 256, 0, stream>>>(emb, emb_h, VV*DD);
  k_percepts<<<BB*TT, DD, 0, stream>>>(idx, emb, pos_emb, ifc_w, ifc_b, ifc_g, ifc_beta, percepts);
  k_recurrence<<<BB, 1024, 0, stream>>>(A);
  k_logits<<<dim3(VV/256, (BB*TT)/16), 128, 0, stream>>>(proj_h, emb_h, out_bias, (float*)d_out);
}
